// Classifier_23003844837400
// MI455X (gfx1250) — matrix operand order repaired and checked by reading, not compiled, not run on hardware
//
#include <hip/hip_runtime.h>

typedef __attribute__((ext_vector_type(16))) _Float16 v16h;
typedef __attribute__((ext_vector_type(8)))  _Float16 v8h;
typedef __attribute__((ext_vector_type(4)))  _Float16 v4h;
typedef __attribute__((ext_vector_type(8)))  float    v8f;

#define BDIM 4
#define IDIM 25
#define JDIM 25
#define MDIM 128
#define NDIM 128
#define KDIM 64

// ---- fragment loads from LDS (row-major [row][k] f16 tiles) ----
// A-matrix 16x32 f16 layout (ISA 7.12.2):
//   lanes 0-15 : M=lane,    VGPR0..3 = K 0..7,  VGPR4..7 = K 16..23
//   lanes16-31 : M=lane-16, VGPR0..3 = K 8..15, VGPR4..7 = K 24..31
__device__ __forceinline__ v16h load_a_frag(const _Float16* row, int hi16, int kstep) {
    const int kbase = (hi16 << 3) + (kstep << 5);
    v8h c0 = *(const v8h*)(row + kbase);        // K = kbase .. kbase+7
    v8h c1 = *(const v8h*)(row + kbase + 16);   // K = kbase+16 .. kbase+23
    return __builtin_shufflevector(c0, c1, 0,1,2,3,4,5,6,7,8,9,10,11,12,13,14,15);
}
// B-matrix 32x16 f16 (same K order as A):
//   lanes 0-15 : N=lane,    VGPR0..3 = K 0..7,  VGPR4..7 = K 16..23
//   lanes16-31 : N=lane-16, VGPR0..3 = K 8..15, VGPR4..7 = K 24..31
// We feed B = S^T, i.e. read row n of the N-major LDS tile.
__device__ __forceinline__ v16h load_b_frag(const _Float16* row, int hi16, int kstep) {
    const int kbase = (hi16 << 3) + (kstep << 5);
    v8h c0 = *(const v8h*)(row + kbase);        // K = kbase .. kbase+7
    v8h c1 = *(const v8h*)(row + kbase + 16);   // K = kbase+16 .. kbase+23
    return __builtin_shufflevector(c0, c1, 0,1,2,3,4,5,6,7,8,9,10,11,12,13,14,15);
}

__device__ __forceinline__ v8f wmma_f16(v16h a, v16h b, v8f c) {
    return __builtin_amdgcn_wmma_f32_16x16x32_f16(false, a, false, b, (short)0, c,
                                                  false, false);
}

__global__ __launch_bounds__(256)
void cosine_relation_kernel(const float* __restrict__ q,
                            const float* __restrict__ s,
                            float* __restrict__ out)
{
    // hi/lo f16 split tiles: 4 * 128*64*2B = 64 KB LDS (WGP has 320 KB)
    __shared__ _Float16 qhi[MDIM * KDIM];
    __shared__ _Float16 qlo[MDIM * KDIM];
    __shared__ _Float16 shi[NDIM * KDIM];
    __shared__ _Float16 slo[NDIM * KDIM];

    const int bij = blockIdx.x;
    const int jj  = bij % JDIM;
    const int ii  = (bij / JDIM) % IDIM;
    const int bb  = bij / (IDIM * JDIM);

    const int tid = threadIdx.x;

    // ---- phase 1: L2-normalize one row per thread, write f16 hi/lo to LDS ----
    {
        const bool isQ = (tid < MDIM);                 // waves 0-3: Q, waves 4-7: S
        const int  row = isQ ? tid : tid - MDIM;
        const float* src = isQ
            ? q + ((size_t)(bb * IDIM + ii) * MDIM + row) * KDIM
            : s + ((size_t)(bb * JDIM + jj) * NDIM + row) * KDIM;

        float ss = 0.f;
        #pragma unroll
        for (int c = 0; c < KDIM; c += 4) {
            float4 v = *(const float4*)(src + c);
            ss += v.x * v.x + v.y * v.y + v.z * v.z + v.w * v.w;
        }
        const float scale = 1.0f / fmaxf(sqrtf(ss), 1e-12f);

        _Float16* dh = (isQ ? qhi : shi) + row * KDIM;
        _Float16* dl = (isQ ? qlo : slo) + row * KDIM;
        #pragma unroll
        for (int c = 0; c < KDIM; c += 4) {
            float4 v = *(const float4*)(src + c);
            float x0 = v.x * scale, x1 = v.y * scale;
            float x2 = v.z * scale, x3 = v.w * scale;
            _Float16 h0 = (_Float16)x0, h1 = (_Float16)x1;
            _Float16 h2 = (_Float16)x2, h3 = (_Float16)x3;
            v4h hh = {h0, h1, h2, h3};
            v4h ll = {(_Float16)(x0 - (float)h0), (_Float16)(x1 - (float)h1),
                      (_Float16)(x2 - (float)h2), (_Float16)(x3 - (float)h3)};
            *(v4h*)(dh + c) = hh;
            *(v4h*)(dl + c) = ll;
        }
    }
    __syncthreads();

    // ---- phase 2: 8 waves x 8 n-tiles of 16x16, split-precision WMMA ----
    const int wave  = tid >> 5;
    const int lane  = tid & 31;
    const int laneN = lane & 15;
    const int hi16  = lane >> 4;
    const int m0    = wave << 4;

    const _Float16* arow_h = qhi + (m0 + laneN) * KDIM;
    const _Float16* arow_l = qlo + (m0 + laneN) * KDIM;
    const v16h a0h = load_a_frag(arow_h, hi16, 0);
    const v16h a0l = load_a_frag(arow_l, hi16, 0);
    const v16h a1h = load_a_frag(arow_h, hi16, 1);
    const v16h a1l = load_a_frag(arow_l, hi16, 1);

    float* const obase = out + (size_t)bij * (MDIM * NDIM);

    #pragma unroll
    for (int nt = 0; nt < NDIM / 16; ++nt) {
        const _Float16* brow_h = shi + (nt * 16 + laneN) * KDIM;
        const _Float16* brow_l = slo + (nt * 16 + laneN) * KDIM;
        const v16h b0h = load_b_frag(brow_h, hi16, 0);
        const v16h b0l = load_b_frag(brow_l, hi16, 0);
        const v16h b1h = load_b_frag(brow_h, hi16, 1);
        const v16h b1l = load_b_frag(brow_l, hi16, 1);

        v8f acc = {};
        // hi*hi + hi*lo + lo*hi recovers ~f32 precision from f16 inputs
        acc = wmma_f16(a0h, b0h, acc);
        acc = wmma_f16(a0h, b0l, acc);
        acc = wmma_f16(a0l, b0h, acc);
        acc = wmma_f16(a1h, b1h, acc);
        acc = wmma_f16(a1h, b1l, acc);
        acc = wmma_f16(a1l, b1h, acc);

        // C/D layout: VGPR r -> M = m0 + r + 8*hi16, N = laneN (+16*nt)
        float* op = obase + (size_t)(m0 + hi16 * 8) * NDIM + nt * 16 + laneN;
        #pragma unroll
        for (int r = 0; r < 8; ++r)
            __builtin_nontemporal_store(acc[r], op + (size_t)r * NDIM);
    }
}

extern "C" void kernel_launch(void* const* d_in, const int* in_sizes, int n_in,
                              void* d_out, int out_size, void* d_ws, size_t ws_size,
                              hipStream_t stream) {
    const float* q = (const float*)d_in[0];
    const float* s = (const float*)d_in[1];
    float* out = (float*)d_out;
    (void)in_sizes; (void)n_in; (void)out_size; (void)d_ws; (void)ws_size;

    dim3 grid(BDIM * IDIM * JDIM);   // 2500 blocks: one per (b,i,j)
    dim3 block(256);                 // 8 wave32 waves
    hipLaunchKernelGGL(cosine_relation_kernel, grid, block, 0, stream, q, s, out);
}
